// SLRTransformer_27573690040963
// MI455X (gfx1250) — matrix operand order repaired and checked by reading, not compiled, not run on hardware
//
#include <hip/hip_runtime.h>
#include <hip/hip_bf16.h>

namespace {

constexpr int Lc  = 6;
constexpr int Bc  = 8;
constexpr int Sc  = 2048;
constexpr int Dc  = 1024;
constexpr int DFF = 4096;
constexpr int Hc  = 16;
constexpr int HDc = Dc / Hc;     // 64
constexpr int Cc  = 100;
constexpr int MTOK = Bc * Sc;    // 16384 token rows

typedef __attribute__((ext_vector_type(16))) __bf16 v16bf;
typedef __attribute__((ext_vector_type(8)))  __bf16 v8bf;
typedef __attribute__((ext_vector_type(8)))  float  v8f;

// ---------------- WMMA GEMM: C[M,N] = op(A[M,K] @ W[K,N] (+bias)) ----------------
// flags: 1 = add bias, 2 = relu, 4 = permute output rows (b*S+s) -> (s*B+b)
constexpr int BM = 128, BN = 128, BK = 32;
constexpr int TSTR = 40;   // LDS row stride (bf16 elems): 80B, multiple of 16B

// A-fragment (16x32 bf16), per ISA 7.12.2 16-bit A layout:
// lane<16: elems 0..7 -> K 0..7, elems 8..15 -> K 16..23
// lane>=16: elems 0..7 -> K 8..15, elems 8..15 -> K 24..31
__device__ inline v16bf frag_a(const __bf16* tile, int lane) {
  const int r  = lane & 15;
  const int ko = (lane >> 4) << 3;           // 0 or 8
  const __bf16* p = tile + r * TSTR;
  v8bf lo = *(const v8bf*)(p + ko);
  v8bf hi = *(const v8bf*)(p + ko + 16);
  v16bf f;
#pragma unroll
  for (int i = 0; i < 8; ++i) { f[i] = lo[i]; f[i + 8] = hi[i]; }
  return f;
}

// B-fragment (32x16 bf16), tile stored n-major [n][k]:
// lane<16: elems 0..7 -> K 0..7, elems 8..15 -> K 16..23; lane>=16: K 8..15, K 24..31
__device__ inline v16bf frag_b(const __bf16* tile, int lane) {
  const int n  = lane & 15;
  const int ko = (lane >> 4) << 3;           // 0 or 8
  const __bf16* p = tile + n * TSTR + ko;
  v8bf lo = *(const v8bf*)(p);
  v8bf hi = *(const v8bf*)(p + 16);
  v16bf f;
#pragma unroll
  for (int i = 0; i < 8; ++i) { f[i] = lo[i]; f[i + 8] = hi[i]; }
  return f;
}

template <bool GUARD>
__global__ __launch_bounds__(256)
void gemm_bf16_wmma(const float* __restrict__ A, const float* __restrict__ W,
                    const float* __restrict__ bias, float* __restrict__ C,
                    int M, int N, int K, int flags) {
  __shared__ __align__(16) __bf16 As[2][BM][TSTR];
  __shared__ __align__(16) __bf16 Bs[2][BN][TSTR];

  const int t    = threadIdx.x;
  const int lane = t & 31;
  const int wid  = t >> 5;
  const int wm   = (wid & 1) * 64;    // wave row offset in tile (2 waves on M)
  const int wn   = (wid >> 1) * 32;   // wave col offset in tile (4 waves on N)
  const int tileM = blockIdx.y * BM;
  const int tileN = blockIdx.x * BN;

  // per-thread global-fetch coordinates
  const int arow = t >> 3;            // 0..31 (+32 per it)
  const int acol = (t & 7) << 2;      // 0..28
  const int brow = t >> 5;            // 0..7  (+8 per it)
  const int bcol = (t & 31) << 2;     // 0..124

  float ar[16], br[16];               // staged next tile (A: 4x float4, B: 4x4)

  auto fetchA = [&](int k0) {
#pragma unroll
    for (int it = 0; it < 4; ++it) {
      const int gr = tileM + arow + (it << 5);
      if (!GUARD || gr < M) {
        const float* p = A + (size_t)gr * K + k0 + acol;
        ar[4 * it + 0] = p[0]; ar[4 * it + 1] = p[1];
        ar[4 * it + 2] = p[2]; ar[4 * it + 3] = p[3];
      } else {
        ar[4 * it + 0] = 0.f; ar[4 * it + 1] = 0.f;
        ar[4 * it + 2] = 0.f; ar[4 * it + 3] = 0.f;
      }
    }
  };
  auto fetchB = [&](int k0) {
#pragma unroll
    for (int it = 0; it < 4; ++it) {
      const int gk = k0 + brow + (it << 3);
      const float* p = W + (size_t)gk * N + tileN + bcol;
#pragma unroll
      for (int j = 0; j < 4; ++j)
        br[4 * it + j] = (!GUARD || (tileN + bcol + j) < N) ? p[j] : 0.f;
    }
  };
  auto stash = [&](int buf) {
#pragma unroll
    for (int it = 0; it < 4; ++it) {
      const int r = arow + (it << 5);
      As[buf][r][acol + 0] = (__bf16)ar[4 * it + 0];
      As[buf][r][acol + 1] = (__bf16)ar[4 * it + 1];
      As[buf][r][acol + 2] = (__bf16)ar[4 * it + 2];
      As[buf][r][acol + 3] = (__bf16)ar[4 * it + 3];
    }
#pragma unroll
    for (int it = 0; it < 4; ++it) {
      const int kk = brow + (it << 3);
#pragma unroll
      for (int j = 0; j < 4; ++j)
        Bs[buf][bcol + j][kk] = (__bf16)br[4 * it + j];
    }
  };

  v8f acc[4][2] = {};

  auto compute = [&](int buf) {
    v16bf af[4], bfr[2];
#pragma unroll
    for (int mi = 0; mi < 4; ++mi) af[mi] = frag_a(&As[buf][wm + mi * 16][0], lane);
#pragma unroll
    for (int ni = 0; ni < 2; ++ni) bfr[ni] = frag_b(&Bs[buf][wn + ni * 16][0], lane);
#pragma unroll
    for (int mi = 0; mi < 4; ++mi)
#pragma unroll
      for (int ni = 0; ni < 2; ++ni)
        acc[mi][ni] = __builtin_amdgcn_wmma_f32_16x16x32_bf16(
            false, af[mi], false, bfr[ni], (short)0, acc[mi][ni], false, false);
  };

  // ---- software pipeline: fetch k+1 while WMMAs consume k (ping-pong LDS) ----
  fetchA(0); fetchB(0);
  stash(0);
  __syncthreads();
  int cur = 0;
  for (int k0 = BK; k0 < K; k0 += BK) {
    if (!GUARD && k0 + BK < K) {  // CDNA5 global_prefetch two tiles ahead
      __builtin_prefetch(A + (size_t)(tileM + arow) * K + k0 + BK + acol, 0, 2);
      __builtin_prefetch(W + (size_t)(k0 + BK + brow) * N + tileN + bcol, 0, 2);
    }
    fetchA(k0); fetchB(k0);   // global loads in flight...
    compute(cur);             // ...hidden behind 8 WMMAs
    stash(cur ^ 1);           // wait + cvt + LDS store into the other buffer
    __syncthreads();
    cur ^= 1;
  }
  compute(cur);

  // ---- epilogue: C/D layout -> lane holds (m = j + 8*(lane>=16), n = lane&15) ----
  const int nloc = lane & 15;
  const int mhi  = (lane >> 4) * 8;
  const bool hasBias = flags & 1;
  const bool doRelu  = flags & 2;
  const bool swapBS  = flags & 4;
#pragma unroll
  for (int mi = 0; mi < 4; ++mi) {
#pragma unroll
    for (int ni = 0; ni < 2; ++ni) {
      const int gc = tileN + wn + ni * 16 + nloc;
      if (GUARD && gc >= N) continue;
      const float bv = hasBias ? bias[gc] : 0.f;
#pragma unroll
      for (int j = 0; j < 8; ++j) {
        const int gr = tileM + wm + mi * 16 + mhi + j;
        if (GUARD && gr >= M) continue;
        float x = acc[mi][ni][j] + bv;
        if (doRelu) x = fmaxf(x, 0.f);
        size_t orow = (size_t)gr;
        if (swapBS) orow = (size_t)(gr % Sc) * Bc + (gr / Sc);
        C[orow * N + gc] = x;
      }
    }
  }
}

// ---------------- LayerNorm of (in1 + in2) over last dim D ----------------
__global__ __launch_bounds__(256)
void ln2_kernel(const float* __restrict__ in1, const float* __restrict__ in2,
                const float* __restrict__ g, const float* __restrict__ b,
                float* __restrict__ out, int bcast1) {
  __shared__ float red[256];
  const int t = threadIdx.x;
  const size_t r = blockIdx.x;
  const float* p1 = in1 + (bcast1 ? 0 : r) * Dc;
  const float* p2 = in2 + r * Dc;
  float loc[4];
  float s = 0.f;
#pragma unroll
  for (int i = 0; i < 4; ++i) { const int c = t + (i << 8); loc[i] = p1[c] + p2[c]; s += loc[i]; }
  red[t] = s; __syncthreads();
  for (int off = 128; off > 0; off >>= 1) { if (t < off) red[t] += red[t + off]; __syncthreads(); }
  const float mean = red[0] * (1.f / Dc);
  __syncthreads();
  s = 0.f;
#pragma unroll
  for (int i = 0; i < 4; ++i) { const float d = loc[i] - mean; s += d * d; }
  red[t] = s; __syncthreads();
  for (int off = 128; off > 0; off >>= 1) { if (t < off) red[t] += red[t + off]; __syncthreads(); }
  const float rstd = rsqrtf(red[0] * (1.f / Dc) + 1e-5f);
#pragma unroll
  for (int i = 0; i < 4; ++i) {
    const int c = t + (i << 8);
    out[r * Dc + c] = (loc[i] - mean) * rstd * g[c] + b[c];
  }
}

// --------- per-token head-mixing attention: scores are [H,H]=[16,16] per token ---------
__global__ __launch_bounds__(256)
void attn16_kernel(const float* __restrict__ q, const float* __restrict__ k,
                   const float* __restrict__ v, float* __restrict__ out, int qBroadcast) {
  __shared__ float pr[16][17];
  const size_t p = blockIdx.x;
  const int t = threadIdx.x;
  const float* qp = q + (qBroadcast ? 0 : p) * Dc;
  const float* kp = k + p * Dc;
  const float* vp = v + p * Dc;
  const int i = t >> 4, j = t & 15;
  float s = 0.f;
#pragma unroll
  for (int d = 0; d < HDc; ++d) s += qp[i * HDc + d] * kp[j * HDc + d];
  pr[i][j] = s * 0.125f;  // 1/sqrt(64)
  __syncthreads();
  if (t < 16) {
    float mx = pr[t][0];
    for (int jj = 1; jj < 16; ++jj) mx = fmaxf(mx, pr[t][jj]);
    float e[16]; float sum = 0.f;
    for (int jj = 0; jj < 16; ++jj) { e[jj] = __expf(pr[t][jj] - mx); sum += e[jj]; }
    const float inv = 1.f / sum;
    for (int jj = 0; jj < 16; ++jj) pr[t][jj] = e[jj] * inv;
  }
  __syncthreads();
#pragma unroll
  for (int rr = 0; rr < 4; ++rr) {
    const int idx = t + (rr << 8);
    const int hh = idx >> 6, d = idx & 63;
    float o = 0.f;
    for (int jj = 0; jj < 16; ++jj) o += pr[hh][jj] * vp[jj * HDc + d];
    out[p * Dc + idx] = o;
  }
}

// ---------------- h = x + pos (broadcast over D) ----------------
__global__ __launch_bounds__(256)
void addpos_kernel(const float* __restrict__ x, const float* __restrict__ pos,
                   float* __restrict__ out) {
  const size_t base = (size_t)blockIdx.x * 1024 + (size_t)threadIdx.x * 4;
#pragma unroll
  for (int j = 0; j < 4; ++j) {
    const size_t id = base + j;
    out[id] = x[id] + pos[id & (Dc - 1)];
  }
}

} // namespace

extern "C" void kernel_launch(void* const* d_in, const int* in_sizes, int n_in,
                              void* d_out, int out_size, void* d_ws, size_t ws_size,
                              hipStream_t stream) {
  (void)in_sizes; (void)n_in; (void)out_size; (void)ws_size;
  const float* x        = (const float*)d_in[0];
  const float* pos      = (const float*)d_in[1];
  const float* clsq     = (const float*)d_in[2];
  const float* enc_wq   = (const float*)d_in[3];
  const float* enc_wk   = (const float*)d_in[4];
  const float* enc_wv   = (const float*)d_in[5];
  const float* enc_wo   = (const float*)d_in[6];
  const float* enc_w1   = (const float*)d_in[7];
  const float* enc_b1   = (const float*)d_in[8];
  const float* enc_w2   = (const float*)d_in[9];
  const float* enc_b2   = (const float*)d_in[10];
  const float* enc_ln1g = (const float*)d_in[11];
  const float* enc_ln1b = (const float*)d_in[12];
  const float* enc_ln2g = (const float*)d_in[13];
  const float* enc_ln2b = (const float*)d_in[14];
  const float* dec_wq   = (const float*)d_in[15];
  const float* dec_wk   = (const float*)d_in[16];
  const float* dec_wv   = (const float*)d_in[17];
  const float* dec_wo   = (const float*)d_in[18];
  const float* dec_w1   = (const float*)d_in[19];
  const float* dec_b1   = (const float*)d_in[20];
  const float* dec_w2   = (const float*)d_in[21];
  const float* dec_b2   = (const float*)d_in[22];
  const float* dec_ln1g = (const float*)d_in[23];
  const float* dec_ln1b = (const float*)d_in[24];
  const float* dec_ln2g = (const float*)d_in[25];
  const float* dec_ln2b = (const float*)d_in[26];
  const float* dec_ln3g = (const float*)d_in[27];
  const float* dec_ln3b = (const float*)d_in[28];
  const float* w_out    = (const float*)d_in[29];
  const float* b_out    = (const float*)d_in[30];
  float* out = (float*)d_out;

  float* w = (float*)d_ws;
  const size_t TOK = (size_t)MTOK * Dc;
  float* hbuf = w;              // encoder hidden
  float* dec  = w + 1 * TOK;    // decoder hidden
  float* qbuf = w + 2 * TOK;
  float* kbuf = w + 3 * TOK;
  float* vbuf = w + 4 * TOK;
  float* abuf = w + 5 * TOK;
  float* tbuf = w + 6 * TOK;    // attn-out / h1 / na
  float* xdb  = w + 7 * TOK;    // decoder xd
  float* ffb  = w + 8 * TOK;    // [MTOK, DFF]

  auto gemm = [&](const float* A, const float* Wt, const float* bias, float* C,
                  int M, int N, int K, int flags) {
    dim3 grid((N + BN - 1) / BN, (M + BM - 1) / BM);
    if ((M % BM) == 0 && (N % BN) == 0)
      gemm_bf16_wmma<false><<<grid, dim3(256), 0, stream>>>(A, Wt, bias, C, M, N, K, flags);
    else
      gemm_bf16_wmma<true><<<grid, dim3(256), 0, stream>>>(A, Wt, bias, C, M, N, K, flags);
  };

  const size_t DD = (size_t)Dc * Dc;
  const size_t DF = (size_t)Dc * DFF;

  // h = x + pos
  addpos_kernel<<<MTOK, 256, 0, stream>>>(x, pos, hbuf);

  // ---------------- encoder ----------------
  for (int i = 0; i < Lc; ++i) {
    gemm(hbuf, enc_wq + i * DD, nullptr, qbuf, MTOK, Dc, Dc, 0);
    gemm(hbuf, enc_wk + i * DD, nullptr, kbuf, MTOK, Dc, Dc, 0);
    gemm(hbuf, enc_wv + i * DD, nullptr, vbuf, MTOK, Dc, Dc, 0);
    attn16_kernel<<<MTOK, 256, 0, stream>>>(qbuf, kbuf, vbuf, tbuf, 0);
    gemm(tbuf, enc_wo + i * DD, nullptr, abuf, MTOK, Dc, Dc, 0);
    ln2_kernel<<<MTOK, 256, 0, stream>>>(hbuf, abuf, enc_ln1g + i * Dc, enc_ln1b + i * Dc,
                                         tbuf, 0);                       // h1
    gemm(tbuf, enc_w1 + i * DF, enc_b1 + (size_t)i * DFF, ffb, MTOK, DFF, Dc, 1 | 2);
    gemm(ffb,  enc_w2 + i * DF, enc_b2 + (size_t)i * Dc,  abuf, MTOK, Dc, DFF, 1);
    ln2_kernel<<<MTOK, 256, 0, stream>>>(abuf, tbuf, enc_ln2g + i * Dc, enc_ln2b + i * Dc,
                                         hbuf, 0);                       // h
  }

  // ---------------- decoder ----------------
  for (int i = 0; i < Lc; ++i) {
    const float* dptr = (i == 0) ? clsq : dec;
    const int drows = (i == 0) ? 1 : MTOK;
    ln2_kernel<<<drows, 256, 0, stream>>>(dptr, dptr, dec_ln1g + i * Dc, dec_ln1b + i * Dc,
                                          xdb, 0);                       // xd
    gemm(dptr, dec_wq + i * DD, nullptr, qbuf, drows, Dc, Dc, 0);
    gemm(hbuf, dec_wk + i * DD, nullptr, kbuf, MTOK, Dc, Dc, 0);
    gemm(hbuf, dec_wv + i * DD, nullptr, vbuf, MTOK, Dc, Dc, 0);
    attn16_kernel<<<MTOK, 256, 0, stream>>>(qbuf, kbuf, vbuf, tbuf, (drows == 1) ? 1 : 0);
    gemm(tbuf, dec_wo + i * DD, nullptr, abuf, MTOK, Dc, Dc, 0);
    ln2_kernel<<<MTOK, 256, 0, stream>>>(xdb, abuf, dec_ln2g + i * Dc, dec_ln2b + i * Dc,
                                         tbuf, (drows == 1) ? 1 : 0);    // na
    gemm(tbuf, dec_w1 + i * DF, dec_b1 + (size_t)i * DFF, ffb, MTOK, DFF, Dc, 1 | 2);
    gemm(ffb,  dec_w2 + i * DF, dec_b2 + (size_t)i * Dc,  abuf, MTOK, Dc, DFF, 1);
    ln2_kernel<<<MTOK, 256, 0, stream>>>(tbuf, abuf, dec_ln3g + i * Dc, dec_ln3b + i * Dc,
                                         dec, 0);                        // dec
  }

  // out[S,B,C] = swapaxes(dec,0,1) @ w_out + b_out
  gemm(dec, w_out, b_out, out, MTOK, Cc, Dc, 1 | 4);
}
